// WindowAttention_30520037605562
// MI455X (gfx1250) — matrix operand order repaired and checked by reading, not compiled, not run on hardware
//
#include <hip/hip_runtime.h>

// ---------------- types ----------------
typedef __attribute__((ext_vector_type(16))) __bf16 bf16x16;
typedef __attribute__((ext_vector_type(8)))  __bf16 bf16x8;
typedef __attribute__((ext_vector_type(8)))  float  f32x8;
typedef int v4i_t __attribute__((vector_size(16)));   // matches builtin param type

#define DIM       384
#define HEADS     12
#define HD        32           // head dim
#define NTOK      49
#define NTOK_PAD  64
#define NWIN      64
#define BW        2048                 // B_ = BATCH*NW
#define MTOT      (BW * NTOK)          // 100352 tokens, divisible by 16
#define MTILES    (MTOT / 16)          // 6272 (divisible by 4)
#define NQKVG     1536                 // 1152 qkv + 384 gate columns
#define QK_SCALE  0.17677669529663687f // 32^-0.5

#if defined(__HIP_DEVICE_COMPILE__) && __has_builtin(__builtin_amdgcn_global_load_async_to_lds_b128)
#define HAVE_ASYNC_LDS 1
typedef __attribute__((address_space(1))) v4i_t* g_v4i_p;
typedef __attribute__((address_space(3))) v4i_t* l_v4i_p;
#else
#define HAVE_ASYNC_LDS 0
#endif

// ---------------- scalar conversions ----------------
static __device__ inline __bf16 f2bf(float f) {
  unsigned u = __builtin_bit_cast(unsigned, f);
  unsigned r = u + 0x7FFFu + ((u >> 16) & 1u);     // round-to-nearest-even
  unsigned short h = (unsigned short)(r >> 16);
  return __builtin_bit_cast(__bf16, h);
}
static __device__ inline float bf2f(__bf16 b) {
  unsigned short u = __builtin_bit_cast(unsigned short, b);
  unsigned v = ((unsigned)u) << 16;
  return __builtin_bit_cast(float, v);
}

// ---------------- fragment loaders (per CDNA5 ISA 7.12.2) ----------------
// A (16x32, MxK, bf16): lane L (<16) row M=L holds K 0..7 / 16..23,
// lane L+16 holds K 8..15 / 24..31. Two contiguous 16B loads per lane.
static __device__ inline bf16x16 load_a16x32(const __bf16* base, int ld) {
  const int lane = threadIdx.x & 31;
  const int eff  = lane & 15;
  const int hi   = lane >> 4;
  const __bf16* r = base + (size_t)eff * ld;
  bf16x8 lo = *(const bf16x8*)(r + 8 * hi);
  bf16x8 hh = *(const bf16x8*)(r + 16 + 8 * hi);
  bf16x16 o;
#pragma unroll
  for (int i = 0; i < 8; ++i) { o[i] = lo[i]; o[i + 8] = hh[i]; }
  return o;
}
// B (32x16, KxN, bf16) from a K-major matrix: lane L holds column N=L%16,
// K 0..7 / 16..23 (L<16) or K 8..15 / 24..31 (L>=16), strided loads.
static __device__ inline bf16x16 load_b32x16(const __bf16* base, int ld) {
  const int lane = threadIdx.x & 31;
  const int eff  = lane & 15;
  const int hi   = lane >> 4;
  bf16x16 o;
#pragma unroll
  for (int i = 0; i < 8; ++i) {
    o[i]     = base[(size_t)(8 * hi + i) * ld + eff];
    o[i + 8] = base[(size_t)(16 + 8 * hi + i) * ld + eff];
  }
  return o;
}

#define WMMA_BF16(A, B, C) \
  __builtin_amdgcn_wmma_f32_16x16x32_bf16(false, (A), false, (B), (short)0, (C), false, false)

// ---------------- cooperative panel staging: K x 64 bf16 -> LDS ----------------
// 128 threads copy `rows` x 64 bf16 (row stride ldg in global, 64 in LDS).
// Uses CDNA5 async global->LDS DMA when the toolchain exposes it.
static __device__ inline void stage_panel(const __bf16* __restrict__ g, int ldg,
                                          __bf16* lds, int rows) {
  const int t = threadIdx.x;           // blockDim.x == 128
  const int chunks = rows * 8;         // 16B chunks, 8 per 64-col row
  for (int idx = t; idx < chunks; idx += 128) {
    const int k = idx >> 3, part = idx & 7;
    const __bf16* src = g + (size_t)k * ldg + part * 8;
    __bf16* dst = lds + k * 64 + part * 8;
#if HAVE_ASYNC_LDS
    // flat LDS address: addr[31:0] is the LDS byte offset (ISA 10.2)
    __builtin_amdgcn_global_load_async_to_lds_b128(
        (g_v4i_p)(uintptr_t)src,
        (l_v4i_p)(uint32_t)(uintptr_t)dst,
        0, 0);
#else
    *(bf16x8*)dst = *(const bf16x8*)src;
#endif
  }
#if HAVE_ASYNC_LDS
#if __has_builtin(__builtin_amdgcn_s_wait_asynccnt)
  __builtin_amdgcn_s_wait_asynccnt(0);
#else
  asm volatile("s_wait_asynccnt 0" ::: "memory");
#endif
#endif
  __syncthreads();
}

// ---------------- prep kernels ----------------
__global__ void cvt_x_kernel(const float* __restrict__ x, __bf16* __restrict__ xb, int n) {
  int i = blockIdx.x * blockDim.x + threadIdx.x;
  if (i < n) xb[i] = f2bf(x[i]);
}
// wt[k*1536 + n] = (n<1152 ? qkv_w[n,k] : gate_w[n-1152,k])  (K-major weights)
__global__ void build_wt_kernel(const float* __restrict__ qkvw,
                                const float* __restrict__ gatew,
                                __bf16* __restrict__ wt) {
  int idx = blockIdx.x * blockDim.x + threadIdx.x;
  if (idx < DIM * NQKVG) {
    int k = idx / NQKVG, n = idx % NQKVG;
    float v = (n < 3 * DIM) ? qkvw[n * DIM + k] : gatew[(n - 3 * DIM) * DIM + k];
    wt[idx] = f2bf(v);
  }
}
__global__ void build_pwt_kernel(const float* __restrict__ projw, __bf16* __restrict__ pwt) {
  int idx = blockIdx.x * blockDim.x + threadIdx.x;
  if (idx < DIM * DIM) {
    int k = idx / DIM, n = idx % DIM;
    pwt[idx] = f2bf(projw[n * DIM + k]);
  }
}

// ---------------- kernel 2: fused QKV + gate GEMM ----------------
// 4 waves / block; block stages the 384x64 weight panel once in LDS,
// each wave computes one 16x64 output tile. C[m,n] = sum_k xb[m,k]*wt[k,n]
__global__ __launch_bounds__(128) void qkv_gate_gemm(
    const __bf16* __restrict__ xb, const __bf16* __restrict__ wt,
    const float* __restrict__ qkv_b, const float* __restrict__ gate_b,
    __bf16* __restrict__ qws, __bf16* __restrict__ ktws,
    __bf16* __restrict__ vws, __bf16* __restrict__ gws) {
  __shared__ __attribute__((aligned(32))) __bf16 Blds[DIM * 64];
  const int strip = blockIdx.y * 64;
  stage_panel(wt + strip, NQKVG, Blds, DIM);

  const int wave = threadIdx.x >> 5;
  const int m0 = (blockIdx.x * 4 + wave) * 16;
  f32x8 acc[4] = {{}, {}, {}, {}};
  const __bf16* arow = xb + (size_t)m0 * DIM;
#pragma unroll 4
  for (int kt = 0; kt < DIM / 32; ++kt) {
    if (kt + 1 < DIM / 32) __builtin_prefetch(arow + (kt + 1) * 32, 0, 0);
    bf16x16 a = load_a16x32(arow + kt * 32, DIM);
#pragma unroll
    for (int nt = 0; nt < 4; ++nt) {
      bf16x16 b = load_b32x16(&Blds[(kt * 32) * 64 + nt * 16], 64);
      acc[nt] = WMMA_BF16(a, b, acc[nt]);
    }
  }
  const int lane = threadIdx.x & 31;
  const int eff = lane & 15, hi = lane >> 4;
#pragma unroll
  for (int nt = 0; nt < 4; ++nt) {
    const int n = strip + nt * 16 + eff;
#pragma unroll
    for (int r = 0; r < 8; ++r) {
      const int row = m0 + r + 8 * hi;      // global token index
      const int bb = row / NTOK, t = row % NTOK;
      float val = acc[nt][r];
      if (n < 3 * DIM) {                    // q / k / v
        val += qkv_b[n];
        const int which = n / DIM;
        const int c = n % DIM;
        const int h = c / HD, dd = c % HD;
        const size_t bh = (size_t)(bb * HEADS + h);
        if (which == 0)
          qws[(bh * NTOK_PAD + t) * HD + dd] = f2bf(val * QK_SCALE);
        else if (which == 1)
          ktws[(bh * HD + dd) * NTOK_PAD + t] = f2bf(val);   // K stored d-major
        else
          vws[(bh * NTOK_PAD + t) * HD + dd] = f2bf(val);
      } else {                              // gate: sigmoid
        const int c = n - 3 * DIM;
        float g = 1.0f / (1.0f + __expf(-(val + gate_b[c])));
        gws[(size_t)row * DIM + c] = f2bf(g);
      }
    }
  }
}

// ---------------- kernel 3: windowed attention per (b,h) ----------------
__global__ __launch_bounds__(128) void attn_kernel(
    const __bf16* __restrict__ qws, const __bf16* __restrict__ ktws,
    const __bf16* __restrict__ vws, const __bf16* __restrict__ gws,
    const float* __restrict__ mask, const float* __restrict__ table,
    const int* __restrict__ relidx, __bf16* __restrict__ yws) {
  __shared__ __attribute__((aligned(32))) float  S[NTOK_PAD * NTOK_PAD];
  __shared__ __attribute__((aligned(32))) __bf16 P[NTOK_PAD * NTOK_PAD];
  const int bh = blockIdx.x;
  const int b = bh / HEADS, h = bh % HEADS;
  const int w = b & (NWIN - 1);
  const int wave = threadIdx.x >> 5;
  const int lane = threadIdx.x & 31;
  const int eff = lane & 15, hi = lane >> 4;

  const __bf16* qbase  = qws  + (size_t)bh * NTOK_PAD * HD;
  const __bf16* ktbase = ktws + (size_t)bh * HD * NTOK_PAD;
  const __bf16* vbase  = vws  + (size_t)bh * NTOK_PAD * HD;

  // S = Q K^T : wave m-tile x 4 n-tiles, K dim = 32 (one WMMA each)
  {
    bf16x16 a = load_a16x32(qbase + wave * 16 * HD, HD);
#pragma unroll
    for (int nt = 0; nt < 4; ++nt) {
      f32x8 acc = {};
      bf16x16 bb = load_b32x16(ktbase + nt * 16, NTOK_PAD);  // rows (k) = d, cols = tokens
      acc = WMMA_BF16(a, bb, acc);
#pragma unroll
      for (int r = 0; r < 8; ++r)
        S[(wave * 16 + r + 8 * hi) * NTOK_PAD + nt * 16 + eff] = acc[r];
    }
  }
  __syncthreads();

  // softmax row-wise (+ relative position bias + window mask)
  {
    const int i = threadIdx.x;
    if (i < NTOK_PAD) {
      if (i < NTOK) {
        float m = -3.0e38f;
        for (int j = 0; j < NTOK; ++j) {
          float s = S[i * NTOK_PAD + j]
                  + table[relidx[i * NTOK + j] * HEADS + h]
                  + mask[((size_t)w * NTOK + i) * NTOK + j];
          S[i * NTOK_PAD + j] = s;
          m = fmaxf(m, s);
        }
        float sum = 0.0f;
        for (int j = 0; j < NTOK; ++j) {
          float e = __expf(S[i * NTOK_PAD + j] - m);
          S[i * NTOK_PAD + j] = e;
          sum += e;
        }
        float inv = 1.0f / sum;
        for (int j = 0; j < NTOK_PAD; ++j)
          P[i * NTOK_PAD + j] = (j < NTOK) ? f2bf(S[i * NTOK_PAD + j] * inv) : f2bf(0.0f);
      } else {
        for (int j = 0; j < NTOK_PAD; ++j) P[i * NTOK_PAD + j] = f2bf(0.0f);
      }
    }
  }
  __syncthreads();

  // O = P V : 2 k-tiles x 2 n-tiles per wave
  f32x8 o0 = {}, o1 = {};
#pragma unroll
  for (int kt = 0; kt < 2; ++kt) {
    bf16x16 ap = load_a16x32(&P[wave * 16 * NTOK_PAD + kt * 32], NTOK_PAD);
    bf16x16 b0 = load_b32x16(vbase + (size_t)(kt * 32) * HD + 0, HD);
    bf16x16 b1 = load_b32x16(vbase + (size_t)(kt * 32) * HD + 16, HD);
    o0 = WMMA_BF16(ap, b0, o0);
    o1 = WMMA_BF16(ap, b1, o1);
  }
  // gated epilogue -> bf16 y (token-major, channel = h*32+d)
#pragma unroll
  for (int r = 0; r < 8; ++r) {
    const int t = wave * 16 + r + 8 * hi;
    if (t < NTOK) {
      const size_t base = ((size_t)b * NTOK + t) * DIM + h * HD;
      float g0 = bf2f(gws[base + eff]);
      float g1 = bf2f(gws[base + 16 + eff]);
      yws[base + eff]      = f2bf(o0[r] * g0);
      yws[base + 16 + eff] = f2bf(o1[r] * g1);
    }
  }
}

// ---------------- kernel 4: output projection ----------------
__global__ __launch_bounds__(128) void proj_gemm(
    const __bf16* __restrict__ yws, const __bf16* __restrict__ pwt,
    const float* __restrict__ proj_b, float* __restrict__ out) {
  __shared__ __attribute__((aligned(32))) __bf16 Blds[DIM * 64];
  const int strip = blockIdx.y * 64;
  stage_panel(pwt + strip, DIM, Blds, DIM);

  const int wave = threadIdx.x >> 5;
  const int m0 = (blockIdx.x * 4 + wave) * 16;
  f32x8 acc[4] = {{}, {}, {}, {}};
  const __bf16* arow = yws + (size_t)m0 * DIM;
#pragma unroll 4
  for (int kt = 0; kt < DIM / 32; ++kt) {
    if (kt + 1 < DIM / 32) __builtin_prefetch(arow + (kt + 1) * 32, 0, 0);
    bf16x16 a = load_a16x32(arow + kt * 32, DIM);
#pragma unroll
    for (int nt = 0; nt < 4; ++nt) {
      bf16x16 b = load_b32x16(&Blds[(kt * 32) * 64 + nt * 16], 64);
      acc[nt] = WMMA_BF16(a, b, acc[nt]);
    }
  }
  const int lane = threadIdx.x & 31;
  const int eff = lane & 15, hi = lane >> 4;
#pragma unroll
  for (int nt = 0; nt < 4; ++nt) {
    const int n = strip + nt * 16 + eff;
    const float pb = proj_b[n];
#pragma unroll
    for (int r = 0; r < 8; ++r) {
      const int row = m0 + r + 8 * hi;
      out[(size_t)row * DIM + n] = acc[nt][r] + pb;
    }
  }
}

// ---------------- launch ----------------
extern "C" void kernel_launch(void* const* d_in, const int* in_sizes, int n_in,
                              void* d_out, int out_size, void* d_ws, size_t ws_size,
                              hipStream_t stream) {
  const float* x      = (const float*)d_in[0];
  const float* mask   = (const float*)d_in[1];
  const float* qkv_w  = (const float*)d_in[2];
  const float* qkv_b  = (const float*)d_in[3];
  const float* proj_w = (const float*)d_in[4];
  const float* proj_b = (const float*)d_in[5];
  const float* gate_w = (const float*)d_in[6];
  const float* gate_b = (const float*)d_in[7];
  const float* table  = (const float*)d_in[8];
  const int*   relidx = (const int*)d_in[9];
  float* out = (float*)d_out;

  char* ws = (char*)d_ws;
  size_t off = 0;
  auto alloc = [&](size_t bytes) -> void* {
    void* p = (void*)(ws + off);
    off += (bytes + 255) & ~(size_t)255;
    return p;
  };
  __bf16* xb   = (__bf16*)alloc((size_t)MTOT * DIM * 2);
  __bf16* wt   = (__bf16*)alloc((size_t)DIM * NQKVG * 2);
  __bf16* pwt  = (__bf16*)alloc((size_t)DIM * DIM * 2);
  __bf16* qws  = (__bf16*)alloc((size_t)BW * HEADS * NTOK_PAD * HD * 2);
  __bf16* ktws = (__bf16*)alloc((size_t)BW * HEADS * HD * NTOK_PAD * 2);
  __bf16* vws  = (__bf16*)alloc((size_t)BW * HEADS * NTOK_PAD * HD * 2);
  __bf16* gws  = (__bf16*)alloc((size_t)MTOT * DIM * 2);
  __bf16* yws  = (__bf16*)alloc((size_t)MTOT * DIM * 2);
  (void)ws_size; (void)in_sizes; (void)n_in; (void)out_size;

  const int nx = MTOT * DIM;
  cvt_x_kernel<<<(nx + 255) / 256, 256, 0, stream>>>(x, xb, nx);
  build_wt_kernel<<<(DIM * NQKVG + 255) / 256, 256, 0, stream>>>(qkv_w, gate_w, wt);
  build_pwt_kernel<<<(DIM * DIM + 255) / 256, 256, 0, stream>>>(proj_w, pwt);

  qkv_gate_gemm<<<dim3(MTILES / 4, NQKVG / 64), 128, 0, stream>>>(
      xb, wt, qkv_b, gate_b, qws, ktws, vws, gws);

  attn_kernel<<<BW * HEADS, 128, 0, stream>>>(
      qws, ktws, vws, gws, mask, table, relidx, yws);

  proj_gemm<<<dim3(MTILES / 4, DIM / 64), 128, 0, stream>>>(yws, pwt, proj_b, out);
}
